// C_AffineCouplingLayer_68058051772936
// MI455X (gfx1250) — compile-verified, matrix operand order checked by reading, not run on hardware
//
#include <hip/hip_runtime.h>
#include <math.h>

typedef __attribute__((ext_vector_type(16))) _Float16 v16h;
typedef __attribute__((ext_vector_type(8)))  _Float16 v8h;
typedef __attribute__((ext_vector_type(2)))  _Float16 v2h;
typedef __attribute__((ext_vector_type(8)))  float    v8f;

union HFrag { v16h v; v8h p[2]; };
union F2U  { float2 f; unsigned long long u; };

// LDS layout (halves):
//  sWin  [128*64 ]  @ 0
//  sWb1  [128*128]  @ 8192
//  sWb2  [128*128]  @ 24576
//  sWout [128*128]  @ 40960
//  sBias (float[512]) @ halves 57344  (b_in,b_b1,b_b2,b_out)
//  sAct  8 waves * 16*128 halves
#define SMEM_BYTES (57344*2 + 512*4 + 8*16*128*2)   // 149504 B < 320KB/WGP

__device__ __forceinline__ float elu1(float v) {
    return v > 0.0f ? v : (__expf(v) - 1.0f);
}

__device__ __forceinline__ float fast_tanh(float v) {
#if __has_builtin(__builtin_amdgcn_tanhf)
    return __builtin_amdgcn_tanhf(v);        // gfx1250 V_TANH_F32
#else
    return tanhf(v);
#endif
}

// One layer: D(16x128) = A(16xK) * W^T + bias, f16 WMMA, f32 accumulate.
// A fragments preloaded for all k-steps; B fragments double-buffered so the
// load for step i+1 is issued before the WMMA of step i.
template<int K>
__device__ __forceinline__ void gemm_layer(const _Float16* actRow,
                                           const _Float16* w,
                                           const float* biasv,
                                           int n0, int kh, v8f acc[8]) {
    #pragma unroll
    for (int j = 0; j < 8; ++j) {
        float bv = biasv[j];
        #pragma unroll
        for (int e = 0; e < 8; ++e) acc[j][e] = bv;
    }

    const v8h* arow = (const v8h*)actRow;
    HFrag a[K/32];                           // preload all A fragments
    #pragma unroll
    for (int kk = 0; kk < K/32; ++kk) {
        a[kk].p[0] = arow[4*kk + kh];
        a[kk].p[1] = arow[4*kk + 2 + kh];
    }

    constexpr int NT = 8 * (K/32);           // flattened (kstep, ntile) steps
    HFrag b[2];
    {   // prime: B for step 0 (kk=0, j=n-tile 0)
        const v8h* wrow = (const v8h*)(w + (size_t)n0 * K);
        b[0].p[0] = wrow[kh];
        b[0].p[1] = wrow[2 + kh];
    }
    #pragma unroll
    for (int i = 0; i < NT; ++i) {
        if (i + 1 < NT) {                    // prefetch B for next step
            int kn = (i + 1) >> 3, jn = (i + 1) & 7;
            const v8h* wrow = (const v8h*)(w + (size_t)(16*jn + n0) * K);
            b[(i+1)&1].p[0] = wrow[4*kn + kh];
            b[(i+1)&1].p[1] = wrow[4*kn + 2 + kh];
        }
        acc[i & 7] = __builtin_amdgcn_wmma_f32_16x16x32_f16(
                         false, a[i >> 3].v, false, b[i & 1].v,
                         (short)0, acc[i & 7], false, false);
    }
}

__global__ void __launch_bounds__(256)
coupling_kernel(const float* __restrict__ x,
                const float* __restrict__ Win,  const float* __restrict__ bin,
                const float* __restrict__ Wb1,  const float* __restrict__ bb1,
                const float* __restrict__ Wb2,  const float* __restrict__ bb2,
                const float* __restrict__ Wout, const float* __restrict__ bout,
                float* __restrict__ y, int nRows, int tilesPerWG)
{
    extern __shared__ _Float16 smem[];
    _Float16* sWin  = smem;
    _Float16* sWb1  = smem + 8192;
    _Float16* sWb2  = smem + 24576;
    _Float16* sWout = smem + 40960;
    float*    sBias = (float*)(smem + 57344);
    _Float16* sAct  = (_Float16*)(sBias + 512);

    const int tid  = threadIdx.x;
    const int lane = tid & 31;
    const int wave = tid >> 5;
    const int n0   = lane & 15;   // column / A-row index within tile
    const int kh   = lane >> 4;   // which k-half of the fragment this lane holds

    // ---- cooperative weight conversion f32 -> f16 into LDS (L2-resident) ----
    for (int i = tid; i < 128*64;  i += 256) sWin[i]  = (_Float16)Win[i];
    for (int i = tid; i < 128*128; i += 256) sWb1[i]  = (_Float16)Wb1[i];
    for (int i = tid; i < 128*128; i += 256) sWb2[i]  = (_Float16)Wb2[i];
    for (int i = tid; i < 128*128; i += 256) sWout[i] = (_Float16)Wout[i];
    if (tid < 128) {
        sBias[      tid] = bin [tid];
        sBias[128 + tid] = bb1 [tid];
        sBias[256 + tid] = bb2 [tid];
        sBias[384 + tid] = bout[tid];
    }
    __syncthreads();   // only cross-wave sync needed; act buffers are wave-private

    // hoist per-lane bias values (column n = 16*j + n0) into registers
    float bReg[4][8];
    #pragma unroll
    for (int L = 0; L < 4; ++L)
        #pragma unroll
        for (int j = 0; j < 8; ++j)
            bReg[L][j] = sBias[L*128 + 16*j + n0];

    _Float16* act = sAct + wave * (16*128);          // this wave's activation tile
    const _Float16* myRow = act + n0 * 128;          // lane's A row (m = lane&15)

    for (int t = 0; t < tilesPerWG; ++t) {
        long rowTile = ((long)blockIdx.x * tilesPerWG + t) * 128 + (long)wave * 16;
        if (rowTile + 16 > nRows) continue;

        if (t + 1 < tilesPerWG)  // speculative prefetch of next x tile
            __builtin_prefetch(x + (rowTile + 128) * 128, 0, 1);

        // ---- stage x_pass (even dims) as packed f16 into act[16][0:64] ----
        // Phase 1: batch all global loads (one clause, one wait).
        float2 xs[16];
        #pragma unroll
        for (int i = 0; i < 16; ++i) {
            int flat = lane + 32*i;                  // 16 rows x 32 chunks
            int m = flat >> 5, cc = flat & 31;
            const float4 xv = *(const float4*)(x + (rowTile + m)*128 + 4*cc);
            xs[i].x = xv.x;                          // even dims 4*cc, 4*cc+2
            xs[i].y = xv.z;
        }
        // Phase 2: convert + packed b32 LDS stores.
        #pragma unroll
        for (int i = 0; i < 16; ++i) {
            int flat = lane + 32*i;
            int m = flat >> 5, cc = flat & 31;
            v2h p;
            p.x = (_Float16)xs[i].x;
            p.y = (_Float16)xs[i].y;
            *(v2h*)(act + m*128 + 2*cc) = p;
        }

        v8f acc[8], res[8];

        // L0: a0 = x_pass @ Win^T + b_in ; res = a0 ; act = elu(a0)
        gemm_layer<64>(myRow, sWin, bReg[0], n0, kh, acc);
        #pragma unroll
        for (int j = 0; j < 8; ++j)
            #pragma unroll
            for (int mv = 0; mv < 8; ++mv) {
                float v = acc[j][mv];
                res[j][mv] = v;
                act[(mv + 8*kh)*128 + 16*j + n0] = (_Float16)elu1(v);
            }

        // L1: a1 = elu(a0) @ Wb1^T + b_b1 ; res += a1 ; act = elu(a1)
        gemm_layer<128>(myRow, sWb1, bReg[1], n0, kh, acc);
        #pragma unroll
        for (int j = 0; j < 8; ++j)
            #pragma unroll
            for (int mv = 0; mv < 8; ++mv) {
                float v = acc[j][mv];
                res[j][mv] += v;
                act[(mv + 8*kh)*128 + 16*j + n0] = (_Float16)elu1(v);
            }

        // L2: a2 = elu(a1) @ Wb2^T + b_b2 ; res += a2 ; act = elu(res)
        gemm_layer<128>(myRow, sWb2, bReg[2], n0, kh, acc);
        #pragma unroll
        for (int j = 0; j < 8; ++j)
            #pragma unroll
            for (int mv = 0; mv < 8; ++mv) {
                float r = res[j][mv] + acc[j][mv];
                act[(mv + 8*kh)*128 + 16*j + n0] = (_Float16)elu1(r);
            }

        // L3: st = elu(res) @ Wout^T + b_out   (tiles 0..3 = s_raw, 4..7 = t)
        gemm_layer<128>(myRow, sWout, bReg[3], n0, kh, acc);

        // ---- epilogue: s and t for column c live in the same lane ----
        #pragma unroll
        for (int j = 0; j < 4; ++j) {
            int c = 16*j + n0;                       // coupled column, 0..63
            float2 xe[8];
            #pragma unroll                           // batch the 8 x reloads
            for (int mv = 0; mv < 8; ++mv) {
                long r = rowTile + mv + 8*kh;
                xe[mv] = *(const float2*)(x + r*128 + 2*c);   // (pass, trans)
            }
            #pragma unroll
            for (int mv = 0; mv < 8; ++mv) {
                long r = rowTile + mv + 8*kh;
                float s  = acc[j    ][mv];
                float tt = acc[j + 4][mv];
                float scale = __expf(2.0f * fast_tanh(s));  // exp(S_MAX*tanh(s))
                F2U o;
                o.f.x = xe[mv].x;                    // even dim: passthrough
                o.f.y = xe[mv].y * scale + tt;       // odd dim: affine transform
                // y is a never-re-read stream: non-temporal b64 store
                __builtin_nontemporal_store(o.u,
                    (unsigned long long*)(y + r*128 + 2*c));
            }
        }
    }
}

extern "C" void kernel_launch(void* const* d_in, const int* in_sizes, int n_in,
                              void* d_out, int out_size, void* d_ws, size_t ws_size,
                              hipStream_t stream) {
    const float* x    = (const float*)d_in[0];
    const float* Win  = (const float*)d_in[1];
    const float* bin  = (const float*)d_in[2];
    const float* Wb1  = (const float*)d_in[3];
    const float* bb1  = (const float*)d_in[4];
    const float* Wb2  = (const float*)d_in[5];
    const float* bb2  = (const float*)d_in[6];
    const float* Wout = (const float*)d_in[7];
    const float* bout = (const float*)d_in[8];
    float* y = (float*)d_out;

    int nRows = in_sizes[0] / 128;
    const int tilesPerWG = 4;                       // 512 rows per WG
    int numTiles = (nRows + 127) / 128;
    int grid = (numTiles + tilesPerWG - 1) / tilesPerWG;

    (void)hipFuncSetAttribute((const void*)coupling_kernel,
                              hipFuncAttributeMaxDynamicSharedMemorySize,
                              SMEM_BYTES);
    coupling_kernel<<<grid, 256, SMEM_BYTES, stream>>>(
        x, Win, bin, Wb1, bb1, Wb2, bb2, Wout, bout, y, nRows, tilesPerWG);
}
